// GAT_12017318494626
// MI455X (gfx1250) — compile-verified, matrix operand order checked by reading, not run on hardware
//
#include <hip/hip_runtime.h>
#include <hip/hip_bf16.h>
#include <math.h>

#define Bb 32
#define Nn 128
#define Dd 64
#define Ee (Nn * Nn)
#define EPS 1e-5f

typedef __bf16 bf16_t;
typedef __bf16 v16bf __attribute__((ext_vector_type(16)));
typedef float  v8f   __attribute__((ext_vector_type(8)));

// ---------------------------------------------------------------------------
// WMMA fragment helpers (wave32, 16x16x32 bf16 -> f32)
// A (16x32, MxK): lane L holds row m=L&15; lane-half g=L>>4 adds K offset 8.
//   VGPR v<4: K = 2v + 8g {+0,+1};  v>=4: K = 16 + 2(v-4) + 8g {+0,+1}.
// B (32x16, KxN): symmetric layout, lane holds column n=L&15, K striped as A.
// C/D (16x16 f32): lane holds n=L&15, VGPR r holds row m = r + 8*(L>>4).
// ---------------------------------------------------------------------------

__device__ __forceinline__ int kpair_base(int v, int g) {
    return ((v < 4) ? (2 * v) : (16 + 2 * (v - 4))) + 8 * g;
}

// Load 16x32 A fragment (bf16) from a fp32 row-major 16x64 LDS tile (stride 64).
__device__ __forceinline__ v16bf load_a_frag_f32(const float* t, int lane, int khalf) {
    const int m = lane & 15;
    const int g = (lane >> 4) & 1;
    const float* row = t + m * 64 + khalf * 32;
    v16bf a;
#pragma unroll
    for (int v = 0; v < 8; ++v) {
        const int kb = kpair_base(v, g);
        a[2 * v]     = (bf16_t)row[kb];
        a[2 * v + 1] = (bf16_t)row[kb + 1];
    }
    return a;
}

// Load 32x16 B fragment from bf16 weight stored row-major as W[o][k] (64x64),
// representing B[k][n] = W[otile*16 + n][k]  (i.e. multiply by W^T).
__device__ __forceinline__ v16bf load_b_frag(const bf16_t* w, int lane, int otile, int khalf) {
    const int n = (lane & 15) + otile * 16;
    const int g = (lane >> 4) & 1;
    const bf16_t* row = w + n * 64 + khalf * 32;
    v16bf b;
#pragma unroll
    for (int v = 0; v < 8; ++v) {
        const int kb = kpair_base(v, g);
        b[2 * v]     = row[kb];
        b[2 * v + 1] = row[kb + 1];
    }
    return b;
}

// Scatter a 16x16 f32 C/D fragment into a fp32 row-major 16x64 LDS tile.
__device__ __forceinline__ void store_c_lds(float* t, const v8f& c, int lane, int otile) {
    const int n  = (lane & 15) + otile * 16;
    const int hi = (lane >> 4) & 1;
#pragma unroll
    for (int r = 0; r < 8; ++r) t[(r + 8 * hi) * 64 + n] = c[r];
}

__device__ __forceinline__ v8f wmma_bf16(const v16bf& a, const v16bf& b, const v8f& c) {
    return __builtin_amdgcn_wmma_f32_16x16x32_bf16(false, a, false, b, (short)0, c, false, false);
}

// ---------------------------------------------------------------------------
// Node precompute: z_h = x @ Wh^T ; t_src = z_h @ Wp[:,0:64]^T ;
// t_dst = z_h @ Wp[:,64:128]^T ; asd = z_h . Wa[0:64] ; add = z_h . Wa[64:128]
// One wave handles one 16-node tile. Tiles total = B*N/16 = 256.
// ---------------------------------------------------------------------------
__global__ __launch_bounds__(256) void gat_node_pre_kernel(
    const float* __restrict__ x,   // (B,N,D)
    const float* __restrict__ Wh,  // (D,D)
    const float* __restrict__ Wp,  // (D,3D)
    const float* __restrict__ Wa,  // (1,3D)
    float* __restrict__ zh,        // (B,N,D)
    float* __restrict__ ts,        // (B,N,D)
    float* __restrict__ td,        // (B,N,D)
    float* __restrict__ asd,       // (B,N)
    float* __restrict__ add) {     // (B,N)
    __shared__ bf16_t sWh[64 * 64];
    __shared__ bf16_t sWpS[64 * 64];
    __shared__ bf16_t sWpD[64 * 64];
    __shared__ float  sWaS[64];
    __shared__ float  sWaD[64];
    __shared__ float  sTile[8][16 * 64];

    const int tid = threadIdx.x;
    for (int t = tid; t < 4096; t += 256) {
        const int o = t >> 6, k = t & 63;
        sWh[t]  = (bf16_t)Wh[t];
        sWpS[t] = (bf16_t)Wp[o * 192 + k];
        sWpD[t] = (bf16_t)Wp[o * 192 + 64 + k];
    }
    if (tid < 64) { sWaS[tid] = Wa[tid]; sWaD[tid] = Wa[64 + tid]; }
    __syncthreads();

    const int wave = tid >> 5, lane = tid & 31;
    const int wave_id = blockIdx.x * 8 + wave;
    const int n_waves = gridDim.x * 8;
    const int tiles_per_batch = Nn / 16;              // 8
    const int total_tiles = Bb * tiles_per_batch;     // 256
    float* stage = sTile[wave];
    const int hi = (lane >> 4) & 1;
    const int n_lo = lane & 15;

    for (int tile = wave_id; tile < total_tiles; tile += n_waves) {
        const int b = tile / tiles_per_batch;
        const int node_base = (tile - b * tiles_per_batch) * 16;
        const float* src = x + ((size_t)b * Nn + node_base) * 64;

        {   // stage 16x64 fp32 tile, coalesced float4
            const float4* s4 = (const float4*)src;
            float4* d4 = (float4*)stage;
#pragma unroll
            for (int it = 0; it < 8; ++it) d4[lane + 32 * it] = s4[lane + 32 * it];
        }
        v16bf a0 = load_a_frag_f32(stage, lane, 0);
        v16bf a1 = load_a_frag_f32(stage, lane, 1);

        v8f z[4];
#pragma unroll
        for (int ot = 0; ot < 4; ++ot) {
            v8f c = {};
            c = wmma_bf16(a0, load_b_frag(sWh, lane, ot, 0), c);
            c = wmma_bf16(a1, load_b_frag(sWh, lane, ot, 1), c);
            z[ot] = c;
        }
#pragma unroll
        for (int ot = 0; ot < 4; ++ot) store_c_lds(stage, z[ot], lane, ot);

        {   // write z_h coalesced
            float* dst = zh + ((size_t)b * Nn + node_base) * 64;
            const float4* s4 = (const float4*)stage;
            float4* d4 = (float4*)dst;
#pragma unroll
            for (int it = 0; it < 8; ++it) d4[lane + 32 * it] = s4[lane + 32 * it];
        }
        // attention node dots
        if (lane < 16) {
            const float* zr = stage + lane * 64;
            float ds = 0.f, dd = 0.f;
#pragma unroll
            for (int k = 0; k < 64; ++k) { ds += zr[k] * sWaS[k]; dd += zr[k] * sWaD[k]; }
            asd[(size_t)b * Nn + node_base + lane] = ds;
            add[(size_t)b * Nn + node_base + lane] = dd;
        }
        // t_src / t_dst = z_h @ WpS^T / WpD^T
        v16bf za0 = load_a_frag_f32(stage, lane, 0);
        v16bf za1 = load_a_frag_f32(stage, lane, 1);
#pragma unroll
        for (int ot = 0; ot < 4; ++ot) {
            const int n = n_lo + ot * 16;
            v8f cs = {}, cd = {};
            cs = wmma_bf16(za0, load_b_frag(sWpS, lane, ot, 0), cs);
            cs = wmma_bf16(za1, load_b_frag(sWpS, lane, ot, 1), cs);
            cd = wmma_bf16(za0, load_b_frag(sWpD, lane, ot, 0), cd);
            cd = wmma_bf16(za1, load_b_frag(sWpD, lane, ot, 1), cd);
#pragma unroll
            for (int r = 0; r < 8; ++r) {
                const int m = r + 8 * hi;
                ts[((size_t)b * Nn + node_base + m) * 64 + n] = cs[r];
                td[((size_t)b * Nn + node_base + m) * 64 + n] = cd[r];
            }
        }
    }
}

// ---------------------------------------------------------------------------
// Edge pipeline (the heavy, bandwidth-bound kernel):
//   z_e = edge @ We^T
//   attn[e] = leaky( asd[j] + add[i] + z_e . a_e )
//   u[e]   = z_e @ Wp[:,128:192]^T + t_src[j] + t_dst[i]   (pre-BN z_e_new)
// One wave per 16-edge tile (tiles lie within one destination row i).
// Safe in-place: each tile's rows are fully loaded before being overwritten.
// Output is staged through LDS and copied out as b128 stores with the bias
// rows added from coalesced float4 loads.
// ---------------------------------------------------------------------------
__global__ __launch_bounds__(256) void gat_edge_kernel(
    const float* edge_in,            // (B,E,D)  (may alias u_out)
    const float* __restrict__ We,    // (D,D)
    const float* __restrict__ Wp,    // (D,3D)
    const float* __restrict__ Wa,    // (1,3D)
    const float* __restrict__ t_src, // (B,N,D)
    const float* __restrict__ t_dst, // (B,N,D)
    const float* __restrict__ a_src, // (B,N)
    const float* __restrict__ a_dst, // (B,N)
    float* u_out,                    // (B,E,D)  pre-BN
    float* __restrict__ attn_out) {  // (B,E)    leaky-relu applied
    __shared__ bf16_t sWe[64 * 64];
    __shared__ bf16_t sWpE[64 * 64];
    __shared__ float  sAe[64];
    __shared__ float  sTile[8][16 * 64];

    const int tid = threadIdx.x;
    for (int t = tid; t < 4096; t += 256) {
        const int o = t >> 6, k = t & 63;
        sWe[t]  = (bf16_t)We[t];
        sWpE[t] = (bf16_t)Wp[o * 192 + 128 + k];
    }
    if (tid < 64) sAe[tid] = Wa[128 + tid];
    __syncthreads();

    const int wave = tid >> 5, lane = tid & 31;
    const int wave_id = blockIdx.x * 8 + wave;
    const int n_waves = gridDim.x * 8;
    const int tiles_per_batch = Ee / 16;              // 1024
    const int total_tiles = Bb * tiles_per_batch;     // 32768
    float* stage = sTile[wave];

    for (int tile = wave_id; tile < total_tiles; tile += n_waves) {
        const int b = tile / tiles_per_batch;
        const int e_base = (tile - b * tiles_per_batch) * 16;
        const int i_node = e_base >> 7;      // e = i*128 + j
        const int j_base = e_base & 127;
        const float* src = edge_in + ((size_t)b * Ee + e_base) * 64;

        // prefetch next tile this wave will touch
        if (tile + n_waves < total_tiles) {
            const int t2 = tile + n_waves;
            const int b2 = t2 / tiles_per_batch;
            const int e2 = (t2 - b2 * tiles_per_batch) * 16;
            __builtin_prefetch(edge_in + (((size_t)b2 * Ee + e2) * 64) + lane * 16, 0, 1);
        }

        {   // stage 16x64 fp32 tile, coalesced float4 (b128)
            const float4* s4 = (const float4*)src;
            float4* d4 = (float4*)stage;
#pragma unroll
            for (int it = 0; it < 8; ++it) d4[lane + 32 * it] = s4[lane + 32 * it];
        }
        v16bf a0 = load_a_frag_f32(stage, lane, 0);
        v16bf a1 = load_a_frag_f32(stage, lane, 1);

        // z_e = edge @ We^T  (8 wmma), then park z_e in the stage buffer
        v8f z[4];
#pragma unroll
        for (int ot = 0; ot < 4; ++ot) {
            v8f c = {};
            c = wmma_bf16(a0, load_b_frag(sWe, lane, ot, 0), c);
            c = wmma_bf16(a1, load_b_frag(sWe, lane, ot, 1), c);
            z[ot] = c;
        }
#pragma unroll
        for (int ot = 0; ot < 4; ++ot) store_c_lds(stage, z[ot], lane, ot);

        // attention logits for these 16 edges
        if (lane < 16) {
            const float* zr = stage + lane * 64;
            float dot = 0.f;
#pragma unroll
            for (int k = 0; k < 64; ++k) dot += zr[k] * sAe[k];
            float av = a_src[(size_t)b * Nn + j_base + lane] +
                       a_dst[(size_t)b * Nn + i_node] + dot;
            attn_out[(size_t)b * Ee + e_base + lane] = (av >= 0.f) ? av : 0.01f * av;
        }

        // u = z_e @ WpE^T  (8 wmma); park u in the stage buffer
        v16bf za0 = load_a_frag_f32(stage, lane, 0);
        v16bf za1 = load_a_frag_f32(stage, lane, 1);
        v8f u[4];
#pragma unroll
        for (int ot = 0; ot < 4; ++ot) {
            v8f c = {};
            c = wmma_bf16(za0, load_b_frag(sWpE, lane, ot, 0), c);
            c = wmma_bf16(za1, load_b_frag(sWpE, lane, ot, 1), c);
            u[ot] = c;
        }
#pragma unroll
        for (int ot = 0; ot < 4; ++ot) store_c_lds(stage, u[ot], lane, ot);

        // copy out with fused bias rows: b128 loads of t_src/t_dst, b128 stores
        {
            const float4* s4  = (const float4*)stage;
            const float4* td4 = (const float4*)(t_dst + ((size_t)b * Nn + i_node) * 64);
            const float4* ts4 = (const float4*)(t_src + ((size_t)b * Nn + j_base) * 64);
            float4* d4 = (float4*)(u_out + ((size_t)b * Ee + e_base) * 64);
#pragma unroll
            for (int it = 0; it < 8; ++it) {
                const int idx = lane + 32 * it;   // float4 index in 16x64 tile
                const int m   = idx >> 4;         // row (edge within tile)
                const int c4  = idx & 15;         // float4 column
                float4 v = s4[idx];
                const float4 a = ts4[m * 16 + c4];
                const float4 d = td4[c4];
                v.x += a.x + d.x;
                v.y += a.y + d.y;
                v.z += a.z + d.z;
                v.w += a.w + d.w;
                d4[idx] = v;
            }
        }
    }
}

// ---------------------------------------------------------------------------
// Softmax over sources j (row length 128) + aggregation agg[i] = sum_j p*z_h[j]
// One block per (b, i).
// ---------------------------------------------------------------------------
__global__ __launch_bounds__(128) void gat_softmax_agg_kernel(
    const float* __restrict__ attn, // (B,E), leaky already applied
    const float* __restrict__ zh,   // (B,N,D)
    float* __restrict__ agg) {      // (B,N,D)
    const int b = blockIdx.x >> 7;
    const int i = blockIdx.x & 127;
    __shared__ float p[128];
    __shared__ float red[128];
    const int t = threadIdx.x;
    const float v = attn[((size_t)b * Nn + i) * Nn + t];
    red[t] = v; __syncthreads();
    for (int s = 64; s > 0; s >>= 1) { if (t < s) red[t] = fmaxf(red[t], red[t + s]); __syncthreads(); }
    const float mx = red[0]; __syncthreads();
    const float ex = __expf(v - mx);
    p[t] = ex; red[t] = ex; __syncthreads();
    for (int s = 64; s > 0; s >>= 1) { if (t < s) red[t] += red[t + s]; __syncthreads(); }
    const float inv = 1.0f / red[0]; __syncthreads();
    p[t] *= inv; __syncthreads();
    if (t < 64) {
        float s = 0.f;
        const float* zb = zh + (size_t)b * Nn * 64 + t;
#pragma unroll 4
        for (int j = 0; j < 128; ++j) s += p[j] * zb[j * 64];
        agg[((size_t)b * Nn + i) * 64 + t] = s;
    }
}

// ---------------------------------------------------------------------------
// BatchNorm stats: channels C over (B, D) = 2048 values. One block per channel.
// float4 reads (channel row is 16 float4s per batch).
// ---------------------------------------------------------------------------
__global__ __launch_bounds__(256) void bn_stats_kernel(
    const float* __restrict__ src, int C,
    float* __restrict__ mean, float* __restrict__ rstd) {
    const int c = blockIdx.x;
    const int t = threadIdx.x;
    float s = 0.f, s2 = 0.f;
    for (int idx = t; idx < Bb * 16; idx += 256) {   // 512 float4s
        const int b = idx >> 4, q = idx & 15;
        const float4 v = ((const float4*)(src + ((size_t)b * C + c) * 64))[q];
        s  += v.x + v.y + v.z + v.w;
        s2 += v.x * v.x + v.y * v.y + v.z * v.z + v.w * v.w;
    }
    __shared__ float sh[256], sh2[256];
    sh[t] = s; sh2[t] = s2; __syncthreads();
    for (int st = 128; st > 0; st >>= 1) {
        if (t < st) { sh[t] += sh[t + st]; sh2[t] += sh2[t + st]; }
        __syncthreads();
    }
    if (t == 0) {
        const float m = sh[0] * (1.0f / (Bb * 64.f));
        const float var = sh2[0] * (1.0f / (Bb * 64.f)) - m * m;
        mean[c] = m;
        rstd[c] = rsqrtf(var + EPS);
    }
}

// Edge BN + ELU, in place, float4-vectorized. Channel = edge index e.
__global__ __launch_bounds__(256) void edge_apply_kernel(
    float* u, const float* __restrict__ mean, const float* __restrict__ rstd,
    const float* __restrict__ gamma, const float* __restrict__ beta) {
    const size_t total4 = (size_t)Bb * Ee * 16;       // float4 count
    float4* u4 = (float4*)u;
    for (size_t idx = (size_t)blockIdx.x * blockDim.x + threadIdx.x; idx < total4;
         idx += (size_t)gridDim.x * blockDim.x) {
        const int e = (int)((idx >> 4) & (Ee - 1));
        const float sc = rstd[e] * gamma[e];
        const float sh = beta[e] - mean[e] * sc;
        float4 v = u4[idx];
        v.x = v.x * sc + sh; v.y = v.y * sc + sh;
        v.z = v.z * sc + sh; v.w = v.w * sc + sh;
        v.x = (v.x > 0.f) ? v.x : (__expf(v.x) - 1.f);
        v.y = (v.y > 0.f) ? v.y : (__expf(v.y) - 1.f);
        v.z = (v.z > 0.f) ? v.z : (__expf(v.z) - 1.f);
        v.w = (v.w > 0.f) ? v.w : (__expf(v.w) - 1.f);
        u4[idx] = v;
    }
}

// Node: x_new = elu(x + bn(agg)). Channel = node index i (N = 128).
__global__ __launch_bounds__(256) void node_apply_kernel(
    const float* __restrict__ xin, const float* __restrict__ agg,
    const float* __restrict__ mean, const float* __restrict__ rstd,
    const float* __restrict__ gamma, const float* __restrict__ beta,
    float* __restrict__ xout) {
    const int idx = blockIdx.x * blockDim.x + threadIdx.x;
    if (idx >= Bb * Nn * 64) return;
    const int c = (idx >> 6) & (Nn - 1);
    const float y = (agg[idx] - mean[c]) * rstd[c] * gamma[c] + beta[c] + xin[idx];
    xout[idx] = (y > 0.f) ? y : (__expf(y) - 1.f);
}

// ---------------------------------------------------------------------------
extern "C" void kernel_launch(void* const* d_in, const int* in_sizes, int n_in,
                              void* d_out, int out_size, void* d_ws, size_t ws_size,
                              hipStream_t stream) {
    const float* x    = (const float*)d_in[0];
    const float* edge = (const float*)d_in[1];
    const float* Wh1  = (const float*)d_in[2];
    const float* We1  = (const float*)d_in[3];
    const float* Wp1  = (const float*)d_in[4];
    const float* Wa1  = (const float*)d_in[5];
    const float* Wh2  = (const float*)d_in[6];
    const float* We2  = (const float*)d_in[7];
    const float* Wp2  = (const float*)d_in[8];
    const float* Wa2  = (const float*)d_in[9];
    const float* gv1  = (const float*)d_in[10];
    const float* bv1  = (const float*)d_in[11];
    const float* ge1  = (const float*)d_in[12];
    const float* be1  = (const float*)d_in[13];
    const float* gv2  = (const float*)d_in[14];
    const float* bv2  = (const float*)d_in[15];
    const float* ge2  = (const float*)d_in[16];
    const float* be2  = (const float*)d_in[17];

    float* out_x = (float*)d_out;                        // (B,N,D)
    float* out_e = (float*)d_out + (size_t)Bb * Nn * Dd; // (B,E,D) — also layer-1 scratch

    float* ws = (float*)d_ws;
    size_t off = 0;
    float* zh   = ws + off; off += (size_t)Bb * Nn * Dd;
    float* ts   = ws + off; off += (size_t)Bb * Nn * Dd;
    float* td   = ws + off; off += (size_t)Bb * Nn * Dd;
    float* asd  = ws + off; off += (size_t)Bb * Nn;
    float* add  = ws + off; off += (size_t)Bb * Nn;
    float* attn = ws + off; off += (size_t)Bb * Ee;
    float* agg  = ws + off; off += (size_t)Bb * Nn * Dd;
    float* x1   = ws + off; off += (size_t)Bb * Nn * Dd;
    float* meanE = ws + off; off += Ee;
    float* rstdE = ws + off; off += Ee;
    float* meanN = ws + off; off += Nn;
    float* rstdN = ws + off; off += Nn;

    const int node_elems = Bb * Nn * Dd;

    // ---------------- layer 1 ----------------
    gat_node_pre_kernel<<<32, 256, 0, stream>>>(x, Wh1, Wp1, Wa1, zh, ts, td, asd, add);
    gat_edge_kernel<<<1024, 256, 0, stream>>>(edge, We1, Wp1, Wa1, ts, td, asd, add, out_e, attn);
    gat_softmax_agg_kernel<<<Bb * Nn, 128, 0, stream>>>(attn, zh, agg);
    bn_stats_kernel<<<Ee, 256, 0, stream>>>(out_e, Ee, meanE, rstdE);
    bn_stats_kernel<<<Nn, 256, 0, stream>>>(agg, Nn, meanN, rstdN);
    edge_apply_kernel<<<4096, 256, 0, stream>>>(out_e, meanE, rstdE, ge1, be1);
    node_apply_kernel<<<(node_elems + 255) / 256, 256, 0, stream>>>(x, agg, meanN, rstdN, gv1, bv1, x1);

    // ---------------- layer 2 (edge tensor processed in place in d_out) ----
    gat_node_pre_kernel<<<32, 256, 0, stream>>>(x1, Wh2, Wp2, Wa2, zh, ts, td, asd, add);
    gat_edge_kernel<<<1024, 256, 0, stream>>>(out_e, We2, Wp2, Wa2, ts, td, asd, add, out_e, attn);
    gat_softmax_agg_kernel<<<Bb * Nn, 128, 0, stream>>>(attn, zh, agg);
    bn_stats_kernel<<<Ee, 256, 0, stream>>>(out_e, Ee, meanE, rstdE);
    bn_stats_kernel<<<Nn, 256, 0, stream>>>(agg, Nn, meanN, rstdN);
    edge_apply_kernel<<<4096, 256, 0, stream>>>(out_e, meanE, rstdE, ge2, be2);
    node_apply_kernel<<<(node_elems + 255) / 256, 256, 0, stream>>>(x1, agg, meanN, rstdN, gv2, bv2, out_x);
}
